// ParallelExperts_67199058313743
// MI455X (gfx1250) — matrix operand order repaired and checked by reading, not compiled, not run on hardware
//
#include <hip/hip_runtime.h>
#include <hip/hip_bf16.h>

typedef __attribute__((ext_vector_type(16))) _Float16 v16h;
typedef __attribute__((ext_vector_type(8)))  _Float16 v8h;
typedef __attribute__((ext_vector_type(4)))  _Float16 v4h;
typedef __attribute__((ext_vector_type(8)))  float    v8f;

#define NEXP   16
#define NTOK   8192
#define DIN    1024
#define DOUT   4096
#define TPE    512          // tokens per expert
#define KC     64           // K chunk staged in LDS
#define KST    68           // LDS halves per column (64 + 4 pad -> 136B = 34 dwords, odd*2)

// ------------------------------------------------------------------
// Kernel 1: per-token LayerNorm (over DIN) + scale/shift -> fp16
// one block per token, 256 threads, 4 floats/thread
// ------------------------------------------------------------------
__global__ __launch_bounds__(256) void ln_fp16_kernel(
    const float* __restrict__ x, const float* __restrict__ gamma,
    const float* __restrict__ beta, _Float16* __restrict__ xh)
{
    const int row  = blockIdx.x;
    const int tid  = threadIdx.x;
    const int e    = row >> 9;                       // TPE = 512 tokens/expert
    const float4 v = *(const float4*)(x + (size_t)row * DIN + tid * 4);

    float s = v.x + v.y + v.z + v.w;
    float q = v.x * v.x + v.y * v.y + v.z * v.z + v.w * v.w;
#pragma unroll
    for (int off = 16; off; off >>= 1) {
        s += __shfl_xor(s, off, 32);
        q += __shfl_xor(q, off, 32);
    }
    __shared__ float rs_[8], rq_[8];
    const int wave = tid >> 5, lane = tid & 31;
    if (lane == 0) { rs_[wave] = s; rq_[wave] = q; }
    __syncthreads();
    s = 0.f; q = 0.f;
#pragma unroll
    for (int i = 0; i < 8; ++i) { s += rs_[i]; q += rq_[i]; }

    const float mu   = s * (1.0f / DIN);
    const float var  = q * (1.0f / DIN) - mu * mu;
    const float rstd = rsqrtf(var + 1e-5f);

    const float4 g  = *(const float4*)(gamma + (size_t)e * DIN + tid * 4);
    const float4 bt = *(const float4*)(beta  + (size_t)e * DIN + tid * 4);

    union { _Float16 h[4]; uint2 u; } o;
    o.h[0] = (_Float16)((v.x - mu) * rstd * g.x + bt.x);
    o.h[1] = (_Float16)((v.y - mu) * rstd * g.y + bt.y);
    o.h[2] = (_Float16)((v.z - mu) * rstd * g.z + bt.z);
    o.h[3] = (_Float16)((v.w - mu) * rstd * g.w + bt.w);
    *(uint2*)(xh + (size_t)row * DIN + tid * 4) = o.u;
}

// ------------------------------------------------------------------
// Kernel 2: grouped GEMM (fp16 WMMA, fp32 accum) + bias + tanh-GELU
// grid = NEXP * 4(Mtiles) * 16(Ntiles) = 1024 blocks, 256 thr = 8 waves
// WG tile 128(M) x 256(N); wave tile 64x64 = 4x4 frags of 16x16
// ------------------------------------------------------------------
__global__ __launch_bounds__(256) void moe_gemm_kernel(
    const _Float16* __restrict__ xh, const float* __restrict__ W,
    const float* __restrict__ bvec, float* __restrict__ out)
{
    __shared__ _Float16 lds[2 * 256 * KST];          // 69,632 B, double buffered B tile

    const int tid   = threadIdx.x;
    const int wave  = tid >> 5;
    const int lane  = tid & 31;
    const int lhalf = lane >> 4;                     // 0|1 : K-half of the fragment
    const int l15   = lane & 15;

    const int blk = blockIdx.x;
    const int e   = blk >> 6;
    const int mt  = blk & 3;                         // M fastest -> same-expert co-residency
    const int nt  = (blk >> 2) & 15;

    const int wm = wave & 1;                         // 2 waves in M
    const int wn = wave >> 1;                        // 4 waves in N

    const int rowBase = e * TPE + mt * 128 + wm * 64;
    const int colWG   = nt * 256;
    const int colBase = colWG + wn * 64;

    const float* We = W + (size_t)e * DIN * DOUT;

    // A fragment row pointers (ISA 16-bit A layout: lane<16 -> K{0..7,16..23})
    const _Float16* Arow[4];
#pragma unroll
    for (int mf = 0; mf < 4; ++mf)
        Arow[mf] = xh + (size_t)(rowBase + mf * 16 + l15) * DIN + lhalf * 8;

    v8f acc[4][4];
#pragma unroll
    for (int i = 0; i < 4; ++i)
#pragma unroll
        for (int j = 0; j < 4; ++j) acc[i][j] = v8f{};

    // W -> LDS staging coords: 64 groups of 4 columns x 4 k-pairs
    const int grp  = tid >> 2;                       // 0..63
    const int kp   = tid & 3;                        // 0..3
    const int ncol = grp * 4;

    auto fill = [&](int buf, int kb) {
        _Float16* Lb = lds + buf * (256 * KST);
#pragma unroll
        for (int pass = 0; pass < 8; ++pass) {
            const int k = pass * 8 + kp * 2;
            const float* src = We + (size_t)(kb + k) * DOUT + colWG + ncol;
            const float4 r0 = *(const float4*)src;
            const float4 r1 = *(const float4*)(src + DOUT);
            const float a0[4] = {r0.x, r0.y, r0.z, r0.w};
            const float a1[4] = {r1.x, r1.y, r1.z, r1.w};
#pragma unroll
            for (int c = 0; c < 4; ++c) {
                union { _Float16 h[2]; unsigned u; } pk;
                pk.h[0] = (_Float16)a0[c];           // K even
                pk.h[1] = (_Float16)a1[c];           // K odd (packed pair for B layout)
                *(unsigned*)&Lb[(ncol + c) * KST + k] = pk.u;
            }
        }
    };

    auto compute = [&](int buf, int kb) {
        const _Float16* Lb = lds + buf * (256 * KST);
#pragma unroll
        for (int ks = 0; ks < 2; ++ks) {
            const int k0 = kb + ks * 32;
            union { v16h v; v8h h[2]; } a[4];
#pragma unroll
            for (int mf = 0; mf < 4; ++mf) {
                a[mf].h[0] = *(const v8h*)(Arow[mf] + k0);
                a[mf].h[1] = *(const v8h*)(Arow[mf] + k0 + 16);
            }
            union { v16h v; v4h q[4]; } bf[4];
            const int kk = ks * 32 + lhalf * 8;      // B layout: lane<16 K {0..7,16..23}, lane>=16 K {8..15,24..31}
#pragma unroll
            for (int nf = 0; nf < 4; ++nf) {
                const _Float16* pb = Lb + (wn * 64 + nf * 16 + l15) * KST + kk;
                bf[nf].q[0] = *(const v4h*)(pb);
                bf[nf].q[1] = *(const v4h*)(pb + 4);
                bf[nf].q[2] = *(const v4h*)(pb + 16);
                bf[nf].q[3] = *(const v4h*)(pb + 20);
            }
#pragma unroll
            for (int mf = 0; mf < 4; ++mf)
#pragma unroll
                for (int nf = 0; nf < 4; ++nf)
                    acc[mf][nf] = __builtin_amdgcn_wmma_f32_16x16x32_f16(
                        false, a[mf].v, false, bf[nf].v,
                        (short)0, acc[mf][nf], false, false);
        }
    };

    fill(0, 0);
    __syncthreads();
    for (int it = 0; it < DIN / KC; ++it) {
        const int buf = it & 1;
        if (it + 1 < DIN / KC) fill(buf ^ 1, (it + 1) * KC);  // overlap next-stage loads with WMMA
        compute(buf, it * KC);
        __syncthreads();
    }

    // epilogue: bias + tanh GELU, fp32 out (C layout: M = r + 8*lhalf, N = l15)
#pragma unroll
    for (int nf = 0; nf < 4; ++nf) {
        const int n = colBase + nf * 16 + l15;
        const float bn = bvec[(size_t)e * DOUT + n];
#pragma unroll
        for (int mf = 0; mf < 4; ++mf) {
            const int row0 = rowBase + mf * 16 + lhalf * 8;
#pragma unroll
            for (int r = 0; r < 8; ++r) {
                const float y = acc[mf][nf][r] + bn;
                const float u = 0.7978845608028654f * (y + 0.044715f * y * y * y);
                out[(size_t)(row0 + r) * DOUT + n] = 0.5f * y * (1.0f + tanhf(u));
            }
        }
    }
}

// ------------------------------------------------------------------
extern "C" void kernel_launch(void* const* d_in, const int* in_sizes, int n_in,
                              void* d_out, int out_size, void* d_ws, size_t ws_size,
                              hipStream_t stream) {
    const float* x     = (const float*)d_in[0];
    // d_in[1] = expert_frequency (balanced: T/E each) -- layout is fixed, unused
    const float* gamma = (const float*)d_in[2];
    const float* beta  = (const float*)d_in[3];
    const float* W     = (const float*)d_in[4];
    const float* bias  = (const float*)d_in[5];
    float* out         = (float*)d_out;
    _Float16* xh       = (_Float16*)d_ws;            // NTOK*DIN fp16 = 16.8 MB scratch

    ln_fp16_kernel<<<NTOK, 256, 0, stream>>>(x, gamma, beta, xh);
    moe_gemm_kernel<<<NEXP * 4 * 16, 256, 0, stream>>>(xh, W, bias, out);
}
